// PointNetPlusPlus_84043920048192
// MI455X (gfx1250) — matrix operand order repaired and checked by reading, not compiled, not run on hardware
//
#include <hip/hip_runtime.h>
#include <cstdint>
#include <cstddef>

// ---------------------------------------------------------------------------
// Types / WMMA fragment helpers (CDNA5 gfx1250, wave32)
// ---------------------------------------------------------------------------
typedef _Float16 h16;
typedef __attribute__((ext_vector_type(16))) _Float16     v16h;
typedef __attribute__((ext_vector_type(8)))  float        v8f;
typedef __attribute__((ext_vector_type(8)))  unsigned int v8u;

union FragCast { v8u u; v16h h; };

// A-matrix 16x32 f16 fragment from row-major [M x lda] buffer.
// ISA layout: lanes 0-15 -> M=lane, VGPR0-3 = K 0..7, VGPR4-7 = K 16..23;
//             lanes 16-31 -> same M, K 8..15 and K 24..31.
__device__ __forceinline__ v16h load_frag_a(const h16* A, int lda, int m0, int k0, int lane) {
  const int m    = m0 + (lane & 15);
  const int half = lane >> 4;
  const h16* row = A + (size_t)m * lda + k0 + half * 8;
  FragCast f;
  f.u[0] = *(const unsigned int*)(row + 0);
  f.u[1] = *(const unsigned int*)(row + 2);
  f.u[2] = *(const unsigned int*)(row + 4);
  f.u[3] = *(const unsigned int*)(row + 6);
  f.u[4] = *(const unsigned int*)(row + 16);
  f.u[5] = *(const unsigned int*)(row + 18);
  f.u[6] = *(const unsigned int*)(row + 20);
  f.u[7] = *(const unsigned int*)(row + 22);
  return f.h;
}

// B-matrix 32x16 f16 fragment from *transposed* weights Bt[N][ldb] (row n = out col).
// ISA layout: lanes hold N=lane&15; lanes 0-15 carry K 0..7 and K 16..23, lanes 16-31 carry K 8..15 and K 24..31.
__device__ __forceinline__ v16h load_frag_bt(const h16* Bt, int ldb, int n0, int k0, int lane) {
  const int n    = n0 + (lane & 15);
  const int half = lane >> 4;
  const h16* row = Bt + (size_t)n * ldb + k0 + half * 8;
  FragCast f;
#pragma unroll
  for (int t = 0; t < 8; ++t) f.u[t] = *(const unsigned int*)(row + 2 * (t & 3) + 16 * (t >> 2));
  return f.h;
}

__device__ __forceinline__ v8f wmma_step(v16h a, v16h b, v8f c) {
  return __builtin_amdgcn_wmma_f32_16x16x32_f16(false, a, false, b, (short)0, c, false, false);
}

// D 16x16 f32 layout: VGPR r holds M = r + 8*(lane>>4), N = lane&15.
__device__ __forceinline__ void store_tile_h16(h16* C, int ldc, int m0, int n0, int lane,
                                               v8f acc, const float* bias, bool relu) {
  const int n    = n0 + (lane & 15);
  const int half = lane >> 4;
  const float bv = bias[n];
#pragma unroll
  for (int r = 0; r < 8; ++r) {
    float v = acc[r] + bv;
    if (relu) v = fmaxf(v, 0.0f);
    C[(size_t)(m0 + r + half * 8) * ldc + n] = (h16)v;
  }
}

// ---------------------------------------------------------------------------
// Weight conversion: f32 W[K][N] -> f16 Wt[Npad][Kpad] (transposed, zero-padded)
// ---------------------------------------------------------------------------
__global__ void wconv_kernel(const float* W, int K, int N, h16* Wt, int Kpad, int Npad) {
  int i = blockIdx.x * 256 + threadIdx.x;
  if (i >= Kpad * Npad) return;
  int n = i / Kpad, k = i % Kpad;
  Wt[i] = (h16)((k < K && n < N) ? W[(size_t)k * N + n] : 0.0f);
}

// ---------------------------------------------------------------------------
// Farthest point sampling: one block per cloud, min-dist array in LDS.
// ---------------------------------------------------------------------------
__global__ __launch_bounds__(256) void fps_kernel(const float* pts, int n, int npoint,
                                                  int* idx_out, float* ctr_out) {
  const int cloud = blockIdx.x;
  const int tid   = threadIdx.x;
  pts     += (size_t)cloud * n * 3;
  idx_out += (size_t)cloud * npoint;
  ctr_out += (size_t)cloud * npoint * 3;
  __shared__ float d[4096];
  __shared__ float rv[256];
  __shared__ int   ri[256];
  __shared__ int   lastSh;
  for (int j = tid; j < n; j += 256) d[j] = 1e10f;
  if (tid == 0) lastSh = 0;
  __syncthreads();
  for (int t = 0; t < npoint; ++t) {
    const int last = lastSh;
    const float lx = pts[last*3+0], ly = pts[last*3+1], lz = pts[last*3+2];
    if (tid == 0) {
      idx_out[t] = last;
      ctr_out[t*3+0] = lx; ctr_out[t*3+1] = ly; ctr_out[t*3+2] = lz;
    }
    float bv = -1.0f; int bi = 0;
    for (int j = tid; j < n; j += 256) {
      float dx = pts[j*3+0]-lx, dy = pts[j*3+1]-ly, dz = pts[j*3+2]-lz;
      float nd = fminf(d[j], dx*dx + dy*dy + dz*dz);
      d[j] = nd;
      if (nd > bv) { bv = nd; bi = j; }
    }
    rv[tid] = bv; ri[tid] = bi;
    __syncthreads();
    for (int s = 128; s > 0; s >>= 1) {
      if (tid < s && rv[tid+s] > rv[tid]) { rv[tid] = rv[tid+s]; ri[tid] = ri[tid+s]; }
      __syncthreads();
    }
    if (tid == 0) lastSh = ri[0];
    __syncthreads();
  }
}

// ---------------------------------------------------------------------------
// Ball query: one thread per center, first-K in index order (PointNet++ semantics)
// ---------------------------------------------------------------------------
__global__ void ballquery_kernel(const float* ctr, const float* pts, int S, int N,
                                 float r2, int K, int* nbr) {
  int gid = blockIdx.x * blockDim.x + threadIdx.x;
  if (gid >= 16 * S) return;
  int cloud = gid / S, s = gid % S;
  const float* c = ctr + ((size_t)cloud * S + s) * 3;
  const float* p = pts + (size_t)cloud * N * 3;
  int* out = nbr + ((size_t)cloud * S + s) * K;
  const float cx = c[0], cy = c[1], cz = c[2];
  int cnt = 0;
  for (int j = 0; j < N && cnt < K; ++j) {
    float dx = p[j*3]-cx, dy = p[j*3+1]-cy, dz = p[j*3+2]-cz;
    if (dx*dx + dy*dy + dz*dz <= r2) out[cnt++] = j;
  }
  int pad = (cnt > 0) ? out[0] : 0;
  for (; cnt < K; ++cnt) out[cnt] = pad;
}

// ---------------------------------------------------------------------------
// SA1 fused MLP: block = 4 groups x 32 nbrs = 128 rows; layers 3->64->64->128
// all activations in LDS; max-pool over each group of 32 rows.
// Tile loops use a readfirstlane'd (SGPR) wave id + constant trip counts so no
// EXEC masking surrounds the WMMA ops.
// ---------------------------------------------------------------------------
__global__ __launch_bounds__(256) void sa1_mlp_kernel(const float* pos, const float* p1,
    const int* nbr1, const h16* W1t, const h16* W2t, const h16* W3t,
    const float* b1, const float* b2, const float* b3, h16* x1) {
  __shared__ h16 actA[128*64];
  __shared__ h16 actB[128*128];
  const int cloud = blockIdx.y;
  const int g0    = blockIdx.x * 4;
  const int tid   = threadIdx.x;
  const int lane  = tid & 31;
  const int wave  = __builtin_amdgcn_readfirstlane(tid >> 5);  // SGPR, wave-uniform
  const float* posC = pos + (size_t)cloud*4096*3;
  const float* p1C  = p1  + (size_t)cloud*512*3;
  const int*   nbrC = nbr1 + ((size_t)cloud*512 + g0)*32;
  if (tid < 128) {
    int g = tid >> 5, k = tid & 31;
    int j = nbrC[g*32 + k];
    const float* cc = p1C + (size_t)(g0+g)*3;
    h16* rowp = actA + tid*64;
    rowp[0] = (h16)(posC[j*3+0]-cc[0]);
    rowp[1] = (h16)(posC[j*3+1]-cc[1]);
    rowp[2] = (h16)(posC[j*3+2]-cc[2]);
    for (int c = 3; c < 32; ++c) rowp[c] = (h16)0.0f;
  }
  __syncthreads();
#pragma unroll
  for (int i = 0; i < 4; ++i) {                      // L1: [128x32] @ [32x64], 32 tiles
    const int t = wave + i * 8;
    const int m0 = (t >> 2) << 4, n0 = (t & 3) << 4;
    v8f acc = {};
    acc = wmma_step(load_frag_a(actA, 64, m0, 0, lane), load_frag_bt(W1t, 32, n0, 0, lane), acc);
    store_tile_h16(actB, 128, m0, n0, lane, acc, b1, true);
  }
  __syncthreads();
#pragma unroll
  for (int i = 0; i < 4; ++i) {                      // L2: [128x64] @ [64x64], 32 tiles
    const int t = wave + i * 8;
    const int m0 = (t >> 2) << 4, n0 = (t & 3) << 4;
    v8f acc = {};
#pragma unroll
    for (int k0 = 0; k0 < 64; k0 += 32)
      acc = wmma_step(load_frag_a(actB, 128, m0, k0, lane), load_frag_bt(W2t, 64, n0, k0, lane), acc);
    store_tile_h16(actA, 64, m0, n0, lane, acc, b2, true);
  }
  __syncthreads();
#pragma unroll
  for (int i = 0; i < 8; ++i) {                      // L3: [128x64] @ [64x128], 64 tiles
    const int t = wave + i * 8;
    const int m0 = (t >> 3) << 4, n0 = (t & 7) << 4;
    v8f acc = {};
#pragma unroll
    for (int k0 = 0; k0 < 64; k0 += 32)
      acc = wmma_step(load_frag_a(actA, 64, m0, k0, lane), load_frag_bt(W3t, 64, n0, k0, lane), acc);
    store_tile_h16(actB, 128, m0, n0, lane, acc, b3, true);
  }
  __syncthreads();
  for (int o = tid; o < 512; o += 256) {             // max over 32 neighbors
    int g = o >> 7, c = o & 127;
    float m = -1e30f;
    for (int r = 0; r < 32; ++r) m = fmaxf(m, (float)actB[(g*32+r)*128 + c]);
    x1[((size_t)cloud*512 + g0 + g)*128 + c] = (h16)m;
  }
}

// ---------------------------------------------------------------------------
// SA2 fused MLP: block = 1 group x 64 nbrs = 64 rows; layers 131(->160)->128->128->256
// ---------------------------------------------------------------------------
__global__ __launch_bounds__(256) void sa2_mlp_kernel(const float* p1, const h16* x1,
    const float* p2, const int* nbr2, const h16* W1t, const h16* W2t, const h16* W3t,
    const float* b1, const float* b2, const float* b3, h16* x2) {
  __shared__ h16 actA[64*160];
  __shared__ h16 actB[64*256];
  const int cloud = blockIdx.y;
  const int g     = blockIdx.x;      // 0..127
  const int tid   = threadIdx.x;
  const int lane  = tid & 31;
  const int wave  = __builtin_amdgcn_readfirstlane(tid >> 5);
  {
    int r = tid >> 2, q = tid & 3;   // 4 threads per row, 40 cols each
    int j = nbr2[((size_t)cloud*128 + g)*64 + r];
    const h16* xs = x1 + ((size_t)cloud*512 + j)*128;
    const float* pj = p1 + ((size_t)cloud*512 + j)*3;
    const float* cc = p2 + ((size_t)cloud*128 + g)*3;
    h16* rowp = actA + r*160;
    for (int c = q*40; c < q*40 + 40; ++c) {
      h16 v;
      if      (c < 128) v = xs[c];
      else if (c == 128) v = (h16)(pj[0]-cc[0]);
      else if (c == 129) v = (h16)(pj[1]-cc[1]);
      else if (c == 130) v = (h16)(pj[2]-cc[2]);
      else v = (h16)0.0f;
      rowp[c] = v;
    }
  }
  __syncthreads();
#pragma unroll
  for (int i = 0; i < 4; ++i) {                      // L1: [64x160] @ [160x128], 32 tiles
    const int t = wave + i * 8;
    const int m0 = (t >> 3) << 4, n0 = (t & 7) << 4;
    v8f acc = {};
#pragma unroll
    for (int k0 = 0; k0 < 160; k0 += 32)
      acc = wmma_step(load_frag_a(actA, 160, m0, k0, lane), load_frag_bt(W1t, 160, n0, k0, lane), acc);
    store_tile_h16(actB, 256, m0, n0, lane, acc, b1, true);
  }
  __syncthreads();
#pragma unroll
  for (int i = 0; i < 4; ++i) {                      // L2: [64x128] @ [128x128], 32 tiles
    const int t = wave + i * 8;
    const int m0 = (t >> 3) << 4, n0 = (t & 7) << 4;
    v8f acc = {};
#pragma unroll
    for (int k0 = 0; k0 < 128; k0 += 32)
      acc = wmma_step(load_frag_a(actB, 256, m0, k0, lane), load_frag_bt(W2t, 128, n0, k0, lane), acc);
    store_tile_h16(actA, 160, m0, n0, lane, acc, b2, true);
  }
  __syncthreads();
#pragma unroll
  for (int i = 0; i < 8; ++i) {                      // L3: [64x128] @ [128x256], 64 tiles
    const int t = wave + i * 8;
    const int m0 = (t >> 4) << 4, n0 = (t & 15) << 4;
    v8f acc = {};
#pragma unroll
    for (int k0 = 0; k0 < 128; k0 += 32)
      acc = wmma_step(load_frag_a(actA, 160, m0, k0, lane), load_frag_bt(W3t, 128, n0, k0, lane), acc);
    store_tile_h16(actB, 256, m0, n0, lane, acc, b3, true);
  }
  __syncthreads();
  {                                                   // max over 64 neighbors
    int c = tid;                                      // 256 cols, 256 threads
    float m = -1e30f;
    for (int r = 0; r < 64; ++r) m = fmaxf(m, (float)actB[r*256 + c]);
    x2[((size_t)cloud*128 + g)*256 + c] = (h16)m;
  }
}

// ---------------------------------------------------------------------------
// SA3 group-all feature build: feat = [x2 | p2 - mean(p2)] padded 259->288
// ---------------------------------------------------------------------------
__global__ __launch_bounds__(256) void feat3_kernel(const float* p2, const h16* x2, h16* feat3) {
  const int cloud = blockIdx.x;
  const int tid   = threadIdx.x;
  __shared__ float mean[3];
  if (tid < 3) {
    float s = 0.0f;
    for (int i = 0; i < 128; ++i) s += p2[((size_t)cloud*128 + i)*3 + tid];
    mean[tid] = s * (1.0f/128.0f);
  }
  __syncthreads();
  for (int o = tid; o < 128*288; o += 256) {
    int r = o / 288, c = o % 288;
    float v;
    if      (c < 256) v = (float)x2[((size_t)cloud*128 + r)*256 + c];
    else if (c < 259) v = p2[((size_t)cloud*128 + r)*3 + (c-256)] - mean[c-256];
    else v = 0.0f;
    feat3[((size_t)cloud*128 + r)*288 + c] = (h16)v;
  }
}

// ---------------------------------------------------------------------------
// Generic WMMA GEMM: out = relu?(A[MxKpad] * Wt^T + bias); one 16x16 tile/wave
// ---------------------------------------------------------------------------
__global__ __launch_bounds__(256) void gemm_kernel(const h16* A, int lda, const h16* Bt, int ldb,
    const float* bias, h16* outH, int ldo, float* outF, int ldof,
    int M, int Npad, int Kpad, int Nreal, int relu) {
  const int lane = threadIdx.x & 31;
  const int wave = __builtin_amdgcn_readfirstlane(threadIdx.x >> 5);
  const int nt    = Npad >> 4;
  const int total = (M >> 4) * nt;
  const int tile  = blockIdx.x * 8 + wave;            // scalar -> scalar branch below
  if (tile >= total) return;
  const int m0 = (tile / nt) << 4;
  const int n0 = (tile % nt) << 4;
  v8f acc = {};
  for (int k0 = 0; k0 < Kpad; k0 += 32)
    acc = wmma_step(load_frag_a(A, lda, m0, k0, lane), load_frag_bt(Bt, ldb, n0, k0, lane), acc);
  const int n    = n0 + (lane & 15);
  const int half = lane >> 4;
  const float bv = (n < Nreal) ? bias[n] : 0.0f;
#pragma unroll
  for (int r = 0; r < 8; ++r) {
    float v = acc[r] + bv;
    if (relu) v = fmaxf(v, 0.0f);
    int m = m0 + r + half * 8;
    if (outH) outH[(size_t)m*ldo + n] = (h16)v;
    if (outF && n < Nreal) outF[(size_t)m*ldof + n] = v;
  }
}

// Max over `rows` rows per cloud (SA3 global pooling)
__global__ void maxpool_rows_kernel(const h16* a, int rows, int cols, int nclouds, h16* out) {
  int o = blockIdx.x * blockDim.x + threadIdx.x;
  if (o >= nclouds * cols) return;
  int cloud = o / cols, c = o % cols;
  float m = -1e30f;
  for (int r = 0; r < rows; ++r) m = fmaxf(m, (float)a[((size_t)cloud*rows + r)*cols + c]);
  out[o] = (h16)m;
}

// ---------------------------------------------------------------------------
// Host orchestration
// ---------------------------------------------------------------------------
extern "C" void kernel_launch(void* const* d_in, const int* in_sizes, int n_in,
                              void* d_out, int out_size, void* d_ws, size_t ws_size,
                              hipStream_t stream) {
  (void)in_sizes; (void)n_in; (void)out_size; (void)ws_size;
  const float* pos = (const float*)d_in[0];
  const float* sa1w[3] = {(const float*)d_in[1],  (const float*)d_in[2],  (const float*)d_in[3]};
  const float* sa1b[3] = {(const float*)d_in[4],  (const float*)d_in[5],  (const float*)d_in[6]};
  const float* sa2w[3] = {(const float*)d_in[7],  (const float*)d_in[8],  (const float*)d_in[9]};
  const float* sa2b[3] = {(const float*)d_in[10], (const float*)d_in[11], (const float*)d_in[12]};
  const float* sa3w[3] = {(const float*)d_in[13], (const float*)d_in[14], (const float*)d_in[15]};
  const float* sa3b[3] = {(const float*)d_in[16], (const float*)d_in[17], (const float*)d_in[18]};
  const float* hw[3]   = {(const float*)d_in[19], (const float*)d_in[20], (const float*)d_in[21]};
  const float* hb[3]   = {(const float*)d_in[22], (const float*)d_in[23], (const float*)d_in[24]};

  char* wp = (char*)d_ws;
  auto alloc = [&](size_t bytes) -> void* {
    void* r = wp; wp += (bytes + 255) & ~(size_t)255; return r;
  };
  // transposed f16 weights [Npad][Kpad]
  h16* W1t1 = (h16*)alloc((size_t)64  * 32   * 2);
  h16* W2t1 = (h16*)alloc((size_t)64  * 64   * 2);
  h16* W3t1 = (h16*)alloc((size_t)128 * 64   * 2);
  h16* W1t2 = (h16*)alloc((size_t)128 * 160  * 2);
  h16* W2t2 = (h16*)alloc((size_t)128 * 128  * 2);
  h16* W3t2 = (h16*)alloc((size_t)256 * 128  * 2);
  h16* W1t3 = (h16*)alloc((size_t)256 * 288  * 2);
  h16* W2t3 = (h16*)alloc((size_t)512 * 256  * 2);
  h16* W3t3 = (h16*)alloc((size_t)1024* 512  * 2);
  h16* W1th = (h16*)alloc((size_t)512 * 1024 * 2);
  h16* W2th = (h16*)alloc((size_t)256 * 512  * 2);
  h16* W3th = (h16*)alloc((size_t)48  * 256  * 2);
  int*   idx1 = (int*)  alloc((size_t)16*512*4);
  float* p1   = (float*)alloc((size_t)16*512*3*4);
  int*   nbr1 = (int*)  alloc((size_t)16*512*32*4);
  int*   idx2 = (int*)  alloc((size_t)16*128*4);
  float* p2   = (float*)alloc((size_t)16*128*3*4);
  int*   nbr2 = (int*)  alloc((size_t)16*128*64*4);
  h16* x1    = (h16*)alloc((size_t)16*512*128*2);
  h16* x2    = (h16*)alloc((size_t)16*128*256*2);
  h16* feat3 = (h16*)alloc((size_t)2048*288*2);
  h16* a3b   = (h16*)alloc((size_t)2048*256*2);
  h16* a3c   = (h16*)alloc((size_t)2048*512*2);
  h16* a3d   = (h16*)alloc((size_t)2048*1024*2);
  h16* gvec  = (h16*)alloc((size_t)16*1024*2);
  h16* h1    = (h16*)alloc((size_t)16*512*2);
  h16* h2    = (h16*)alloc((size_t)16*256*2);

  auto conv = [&](const float* W, int K, int N, h16* Wt, int Kpad, int Npad) {
    int total = Kpad * Npad;
    wconv_kernel<<<(total + 255) / 256, 256, 0, stream>>>(W, K, N, Wt, Kpad, Npad);
  };
  conv(sa1w[0],    3,   64, W1t1,   32,   64);
  conv(sa1w[1],   64,   64, W2t1,   64,   64);
  conv(sa1w[2],   64,  128, W3t1,   64,  128);
  conv(sa2w[0],  131,  128, W1t2,  160,  128);
  conv(sa2w[1],  128,  128, W2t2,  128,  128);
  conv(sa2w[2],  128,  256, W3t2,  128,  256);
  conv(sa3w[0],  259,  256, W1t3,  288,  256);
  conv(sa3w[1],  256,  512, W2t3,  256,  512);
  conv(sa3w[2],  512, 1024, W3t3,  512, 1024);
  conv(hw[0],   1024,  512, W1th, 1024,  512);
  conv(hw[1],    512,  256, W2th,  512,  256);
  conv(hw[2],    256,   40, W3th,  256,   48);

  // ---- SA1 ----
  fps_kernel<<<16, 256, 0, stream>>>(pos, 4096, 512, idx1, p1);
  ballquery_kernel<<<(16*512 + 255)/256, 256, 0, stream>>>(p1, pos, 512, 4096, 0.04f, 32, nbr1);
  sa1_mlp_kernel<<<dim3(128, 16), 256, 0, stream>>>(pos, p1, nbr1, W1t1, W2t1, W3t1,
                                                    sa1b[0], sa1b[1], sa1b[2], x1);
  // ---- SA2 ----
  fps_kernel<<<16, 256, 0, stream>>>(p1, 512, 128, idx2, p2);
  ballquery_kernel<<<(16*128 + 255)/256, 256, 0, stream>>>(p2, p1, 128, 512, 0.16f, 64, nbr2);
  sa2_mlp_kernel<<<dim3(128, 16), 256, 0, stream>>>(p1, x1, p2, nbr2, W1t2, W2t2, W3t2,
                                                    sa2b[0], sa2b[1], sa2b[2], x2);
  // ---- SA3 (group_all) ----
  feat3_kernel<<<16, 256, 0, stream>>>(p2, x2, feat3);
  auto gemm = [&](const h16* A, int lda, const h16* Bt, int ldb, const float* bias,
                  h16* outH, int ldo, float* outF, int ldof,
                  int M, int Npad, int Kpad, int Nreal, int relu) {
    int tiles = (M >> 4) * (Npad >> 4);
    gemm_kernel<<<(tiles + 7) / 8, 256, 0, stream>>>(A, lda, Bt, ldb, bias, outH, ldo,
                                                     outF, ldof, M, Npad, Kpad, Nreal, relu);
  };
  gemm(feat3, 288, W1t3, 288, sa3b[0], a3b, 256, nullptr, 0, 2048, 256, 288, 256, 1);
  gemm(a3b,   256, W2t3, 256, sa3b[1], a3c, 512, nullptr, 0, 2048, 512, 256, 512, 1);
  gemm(a3c,   512, W3t3, 512, sa3b[2], a3d, 1024, nullptr, 0, 2048, 1024, 512, 1024, 1);
  maxpool_rows_kernel<<<(16*1024 + 255)/256, 256, 0, stream>>>(a3d, 128, 1024, 16, gvec);
  // ---- Head ----
  gemm(gvec, 1024, W1th, 1024, hb[0], h1, 512, nullptr, 0, 16, 512, 1024, 512, 1);
  gemm(h1,    512, W2th,  512, hb[1], h2, 256, nullptr, 0, 16, 256,  512, 256, 1);
  gemm(h2,    256, W3th,  256, hb[2], nullptr, 0, (float*)d_out, 40, 16, 48, 256, 40, 0);
}
